// Physics_Attention_Structured_Mesh_3D_21414706938036
// MI455X (gfx1250) — matrix operand order repaired and checked by reading, not compiled, not run on hardware
//
#include <hip/hip_runtime.h>

// ---------------------------------------------------------------------------
// Physics Attention (Structured Mesh 3D) for MI455X / gfx1250 (wave32, WMMA)
// ---------------------------------------------------------------------------
//   K1  cast x (f32) -> Xb (bf16), layout [b][n][256]
//   K2  pack conv_w -> Wb bf16 [tap(27)][cib(8)][co(512)][cik(32)]  (B-frag ready)
//   K3  pack out_w  -> OWb bf16 [cib(16)][co(256)][cik(32)]
//   K4  zero GMAX/GSUM/EID accumulators
//   K5  conv as implicit GEMM (M=131072, N=512, K=6912), bf16 WMMA,
//       128x64 block tile, double-buffered LDS, async-to-LDS B fill,
//       epilogue writes XPb [b][n][512] and XPt [b][h][c][n] (bf16)
//   K6  logits: z = (xp.slice_w^T + slice_b - log(-log(eps))) / temp -> Z,
//       per-(b,h,g) max via ordered-int atomics
//   K7  sum pass: GSUM[g] += exp(z - max)
//   K8  weights pass: SW [b][h][n][g] and SWt [b][h][g][n] bf16
//   K9  eid = sw^T @ xp per (b,h): WMMA, split-K over n with f32 atomics
//   K10 eid finalize: scale by 1/(1+1e-5), transpose to [b][h][c][g] bf16
//   K11 fused: out_x tile (sw @ eid, 8 heads -> LDS 64x512 bf16) then
//       out = out_x @ out_w^T + out_b  (WMMA, K=512)
// ---------------------------------------------------------------------------

typedef unsigned short u16;
typedef __attribute__((ext_vector_type(16))) __bf16 v16bf;
typedef __attribute__((ext_vector_type(8)))  float  v8f;
typedef __attribute__((__vector_size__(16))) int    vi4;   // b128 payload

union FragBF { uint4 q[2]; v16bf v; };

#if __has_builtin(__builtin_amdgcn_global_load_async_to_lds_b128)
#define HAVE_ASYNC 1
#else
#define HAVE_ASYNC 0
#endif

static __device__ __forceinline__ void wait_async0() {
#if HAVE_ASYNC
#if __has_builtin(__builtin_amdgcn_s_wait_asynccnt)
  __builtin_amdgcn_s_wait_asynccnt(0);
#else
  asm volatile("s_wait_asynccnt 0" ::: "memory");
#endif
#endif
}

static __device__ __forceinline__ u16 f2bf(float f) {
  unsigned u = __builtin_bit_cast(unsigned, f);
  u += 0x7FFFu + ((u >> 16) & 1u);            // round-to-nearest-even
  return (u16)(u >> 16);
}
static __device__ __forceinline__ float bf2f(u16 b) {
  return __builtin_bit_cast(float, (unsigned)b << 16);
}
// order-preserving f32 -> u32 encoding for atomic max
static __device__ __forceinline__ unsigned encf(float f) {
  unsigned u = __builtin_bit_cast(unsigned, f);
  return (u & 0x80000000u) ? ~u : (u | 0x80000000u);
}
static __device__ __forceinline__ float decf(unsigned e) {
  unsigned u = (e & 0x80000000u) ? (e & 0x7FFFFFFFu) : ~e;
  return __builtin_bit_cast(float, u);
}
static __device__ __forceinline__ uint4 ld16(const u16* p) {
  return *reinterpret_cast<const uint4*>(p);
}
static __device__ __forceinline__ v8f wmma_bf16(const FragBF& a, const FragBF& b, v8f c) {
  return __builtin_amdgcn_wmma_f32_16x16x32_bf16(false, a.v, false, b.v, (short)0, c,
                                                 false, false);
}

#define NB     4
#define NVOX   32768          // 32*32*32
#define CDIM   256
#define INNER  512
#define NHEAD  8
#define DHEAD  64
#define NGRP   32
#define LNL    2.6257918f     // log(-log(1e-6))

// --------------------------- K1: cast x to bf16 ----------------------------
__global__ void k_cast_x(const float* __restrict__ x, u16* __restrict__ Xb) {
  size_t i = ((size_t)blockIdx.x * 256 + threadIdx.x) * 4;
  float4 v = *reinterpret_cast<const float4*>(x + i);
  unsigned lo = (unsigned)f2bf(v.x) | ((unsigned)f2bf(v.y) << 16);
  unsigned hi = (unsigned)f2bf(v.z) | ((unsigned)f2bf(v.w) << 16);
  *reinterpret_cast<uint2*>(Xb + i) = make_uint2(lo, hi);
}

// ------------------ K2: pack conv_w [co][ci][27] -> Wb ---------------------
// Wb[((tap*8+cib)*512 + co)*32 + cik] = conv_w[co][cib*32+cik][tap]
__global__ void k_pack_w(const float* __restrict__ w, u16* __restrict__ Wb) {
  int o = blockIdx.x * 256 + threadIdx.x;         // < 27*8*512*32
  int cik = o & 31;
  int co  = (o >> 5) & 511;
  int kb  = o >> 14;                              // tap*8 + cib
  int cib = kb & 7, tap = kb >> 3;
  int ci  = cib * 32 + cik;
  Wb[o] = f2bf(w[((size_t)co * CDIM + ci) * 27 + tap]);
}

// ------------------ K3: pack out_w [256][512] -> OWb -----------------------
// OWb[(cib*256 + co)*32 + cik] = out_w[co][cib*32+cik]
__global__ void k_pack_outw(const float* __restrict__ w, u16* __restrict__ OWb) {
  int o = blockIdx.x * 256 + threadIdx.x;         // < 16*256*32
  int cik = o & 31;
  int co  = (o >> 5) & 255;
  int cib = o >> 13;
  OWb[o] = f2bf(w[(size_t)co * INNER + cib * 32 + cik]);
}

// --------------------------- K4: zero accumulators -------------------------
__global__ void k_init(unsigned* __restrict__ GMAX, float* __restrict__ GSUM,
                       float* __restrict__ EID) {
  int i = blockIdx.x * 256 + threadIdx.x;         // 65536 threads
  if (i < 1024) { GMAX[i] = 0u; GSUM[i] = 0.0f; } // enc(-inf) compares lowest
  EID[i] = 0.0f;
}

// --------------------------- K5: conv implicit GEMM ------------------------
// grid (1024 m-tiles, 8 co-tiles), 256 threads (8 waves).
// Block tile: M=128 voxels x N=64 channels; K = 27 taps * 8 blocks of 32.
// Double-buffered LDS; one barrier per k-step; B tile filled async-to-LDS.
__global__ void k_conv(const u16* __restrict__ Xb, const u16* __restrict__ Wb,
                       const float* __restrict__ conv_b,
                       u16* __restrict__ XPb, u16* __restrict__ XPt) {
  __shared__ __align__(16) u16 As[2][128 * 32];   // 16 KB
  __shared__ __align__(16) u16 Bs[2][64 * 32];    //  8 KB

  const int mt  = blockIdx.x;
  const int ct  = blockIdx.y;
  const int t   = threadIdx.x;
  const int m0g = mt * 128;           // flat (b*NVOX + n) base
  const int bb  = m0g >> 15;          // batch
  const int n0  = m0g & (NVOX - 1);
  const int co0 = ct * 64;

  // A-fill: thread -> (voxel fm = t>>1, halfword base fpb = (t&1)*16; +0,+8)
  const int fm  = t >> 1;
  const int fpb = (t & 1) * 16;
  const int nf  = n0 + fm;
  const int vh  = nf >> 10, vw = (nf >> 5) & 31, vd = nf & 31;
  const u16* abase = Xb + ((size_t)(bb * NVOX + nf) * CDIM) + fpb;

  // B-fill: thread -> (co row t>>2, chunk (t&3)*8): one b128 per thread
  const int bro = t >> 2;
  const int bpp = (t & 3) * 8;

  const int lane = t & 31;
  const int wv   = t >> 5;
  const int wm   = wv & 3;            // M quadrant (32 rows)
  const int wn   = wv >> 2;           // N half (32 cols)
  const int lrow = lane & 15;
  const int hi   = lane >> 4;

  v8f acc[2][2] = {};

  // incremental fill state: next (tap offsets, cib) to load
  int fz = -1, fy = -1, fx = -1, fcib = 0, ftap = 0;

  auto do_fill = [&](int buf) {
    // ---- A tile (halo-predicated register loads -> LDS) ----
    {
      const int hh = vh + fz, ww = vw + fy, dd = vd + fx;
      const int shift = fz * 1024 + fy * 32 + fx;       // voxel index delta
      uint4 v0 = make_uint4(0u, 0u, 0u, 0u), v1 = v0;
      if (((unsigned)hh < 32u) && ((unsigned)ww < 32u) && ((unsigned)dd < 32u)) {
        const u16* p = abase + (ptrdiff_t)shift * CDIM + fcib * 32;
        v0 = ld16(p);
        v1 = ld16(p + 8);
      }
      u16* adst = &As[buf][fm * 32 + fpb];
      *reinterpret_cast<uint4*>(adst)     = v0;
      *reinterpret_cast<uint4*>(adst + 8) = v1;
    }
    // ---- B tile (fragment-ready; async DMA to LDS when available) ----
    {
      const size_t bsrc = (((size_t)(ftap * 8 + fcib) * INNER) + co0 + bro) * 32 + bpp;
      u16* bdst = &Bs[buf][bro * 32 + bpp];
#if HAVE_ASYNC
      __builtin_amdgcn_global_load_async_to_lds_b128((vi4*)(Wb + bsrc), (vi4*)bdst,
                                                     0, 0);
#else
      *reinterpret_cast<uint4*>(bdst) = ld16(Wb + bsrc);
#endif
    }
    // ---- advance state ----
    if (++fcib == 8) {
      fcib = 0; ++ftap;
      if (++fx == 2) { fx = -1; if (++fy == 2) { fy = -1; ++fz; } }
    }
  };

  // prologue: fill buffer 0
  do_fill(0);
  wait_async0();
  __syncthreads();

  int p = 0;
  for (int ks = 0; ks < 216; ++ks) {
    if (ks < 215) do_fill(p ^ 1);     // overlap next-tile fill with compute
    FragBF af[2], bfr[2];
    #pragma unroll
    for (int i = 0; i < 2; ++i) {
      const u16* ap = &As[p][(wm * 32 + i * 16 + lrow) * 32] + (hi ? 8 : 0);
      af[i].q[0] = ld16(ap);
      af[i].q[1] = ld16(ap + 16);
      const u16* bp = &Bs[p][(wn * 32 + i * 16 + lrow) * 32] + (hi ? 8 : 0);
      bfr[i].q[0] = ld16(bp);
      bfr[i].q[1] = ld16(bp + 16);
    }
    #pragma unroll
    for (int i = 0; i < 2; ++i)
      #pragma unroll
      for (int j = 0; j < 2; ++j)
        acc[i][j] = wmma_bf16(af[i], bfr[j], acc[i][j]);
    wait_async0();
    __syncthreads();
    p ^= 1;
  }

  // ---- epilogue: + bias, write XPb [b][n][512] and XPt [b][h][c][n] ----
  #pragma unroll
  for (int i = 0; i < 2; ++i) {                 // M tile
    const int nbase = n0 + wm * 32 + i * 16 + hi * 8;   // 8 consecutive n
    #pragma unroll
    for (int j = 0; j < 2; ++j) {               // N tile
      const int co   = co0 + wn * 32 + j * 16 + lrow;
      const float bias = conv_b[co];
      const int hgrp = co >> 6;                 // head
      const int cch  = co & 63;                 // channel within head
      unsigned pk[4];
      #pragma unroll
      for (int e = 0; e < 8; ++e) {
        float v = acc[i][j][e] + bias;
        u16 bv = f2bf(v);
        XPb[((size_t)(bb * NVOX + nbase + e)) * INNER + co] = bv;
        if (e & 1) pk[e >> 1] |= ((unsigned)bv) << 16;
        else       pk[e >> 1]  = (unsigned)bv;
      }
      *reinterpret_cast<uint4*>(
          XPt + ((size_t)((bb * NHEAD + hgrp) * DHEAD + cch)) * NVOX + nbase) =
          make_uint4(pk[0], pk[1], pk[2], pk[3]);
    }
  }
}

// --------------------------- K6: logits + running max ----------------------
__global__ void k_logits(const u16* __restrict__ XPb,
                         const float* __restrict__ slice_w,
                         const float* __restrict__ slice_b,
                         const float* __restrict__ ada_w,
                         const float* __restrict__ ada_b,
                         float* __restrict__ Z, unsigned* __restrict__ GMAX) {
  __shared__ float sw_s[NGRP * DHEAD];
  __shared__ float ada_s[DHEAD];
  __shared__ unsigned smax[NGRP];
  const int bb = blockIdx.z, hg = blockIdx.y;
  const int n  = blockIdx.x * 256 + threadIdx.x;
  for (int i = threadIdx.x; i < NGRP * DHEAD; i += 256) sw_s[i] = slice_w[i];
  if (threadIdx.x < DHEAD) ada_s[threadIdx.x] = ada_w[threadIdx.x];
  if (threadIdx.x < NGRP)  smax[threadIdx.x] = 0u;
  __syncthreads();

  float xv[DHEAD];
  const u16* xp = XPb + ((size_t)(bb * NVOX + n)) * INNER + hg * DHEAD;
  const uint4* xp4 = reinterpret_cast<const uint4*>(xp);
  #pragma unroll
  for (int q = 0; q < 8; ++q) {
    uint4 u = xp4[q];
    unsigned wds[4] = {u.x, u.y, u.z, u.w};
    #pragma unroll
    for (int j = 0; j < 4; ++j) {
      xv[q * 8 + j * 2]     = bf2f((u16)(wds[j] & 0xFFFFu));
      xv[q * 8 + j * 2 + 1] = bf2f((u16)(wds[j] >> 16));
    }
  }
  float ada = ada_b[0];
  #pragma unroll
  for (int c = 0; c < DHEAD; ++c) ada += xv[c] * ada_s[c];
  float temp  = 0.5f + fminf(0.4f, fmaxf(-0.4f, ada));
  float inv_t = 1.0f / temp;

  float* zrow = Z + ((size_t)((bb * NHEAD + hg) * NVOX + n)) * NGRP;
  for (int g = 0; g < NGRP; ++g) {
    float l = slice_b[g] - LNL;
    const float* wr = sw_s + g * DHEAD;
    #pragma unroll
    for (int c = 0; c < DHEAD; ++c) l += xv[c] * wr[c];
    float z = l * inv_t;
    zrow[g] = z;
    atomicMax(&smax[g], encf(z));
  }
  __syncthreads();
  if (threadIdx.x < NGRP)
    atomicMax(&GMAX[(bb * NHEAD + hg) * NGRP + threadIdx.x], smax[threadIdx.x]);
}

// --------------------------- K7: exp-sum pass ------------------------------
__global__ void k_sum(const float* __restrict__ Z, const unsigned* __restrict__ GMAX,
                      float* __restrict__ GSUM) {
  __shared__ float ssum[NGRP];
  const int bb = blockIdx.z, hg = blockIdx.y;
  const int bh = bb * NHEAD + hg;
  const int n  = blockIdx.x * 256 + threadIdx.x;
  if (threadIdx.x < NGRP) ssum[threadIdx.x] = 0.0f;
  __syncthreads();
  const float* zrow = Z + ((size_t)bh * NVOX + n) * NGRP;
  for (int g = 0; g < NGRP; ++g) {
    float m = decf(GMAX[bh * NGRP + g]);
    atomicAdd(&ssum[g], __expf(zrow[g] - m));
  }
  __syncthreads();
  if (threadIdx.x < NGRP)
    atomicAdd(&GSUM[bh * NGRP + threadIdx.x], ssum[threadIdx.x]);
}

// --------------------------- K8: softmax weights ---------------------------
__global__ void k_weights(const float* __restrict__ Z, const unsigned* __restrict__ GMAX,
                          const float* __restrict__ GSUM,
                          u16* __restrict__ SW, u16* __restrict__ SWt) {
  const int bb = blockIdx.z, hg = blockIdx.y;
  const int bh = bb * NHEAD + hg;
  const int n  = blockIdx.x * 256 + threadIdx.x;
  const float* zrow = Z + ((size_t)bh * NVOX + n) * NGRP;
  for (int g = 0; g < NGRP; ++g) {
    float m = decf(GMAX[bh * NGRP + g]);
    float s = GSUM[bh * NGRP + g];
    u16 bv = f2bf(__expf(zrow[g] - m) / s);
    SW[((size_t)bh * NVOX + n) * NGRP + g] = bv;
    SWt[((size_t)bh * NGRP + g) * NVOX + n] = bv;
  }
}

// --------------------------- K9: eid = sw^T @ xp ---------------------------
// grid (32 n-chunks, 32 bh); one wave per block; split-K via f32 atomics.
__global__ void k_eid(const u16* __restrict__ SWt, const u16* __restrict__ XPt,
                      float* __restrict__ EID) {
  const int chunk = blockIdx.x;
  const int bh    = blockIdx.y;
  const int lane  = threadIdx.x;
  const int lrow  = lane & 15, hi = lane >> 4;
  const int n00   = chunk * 1024;
  v8f acc[2][4] = {};
  for (int kk = 0; kk < 32; ++kk) {
    const int n0 = n00 + kk * 32;
    FragBF af[2];
    #pragma unroll
    for (int m = 0; m < 2; ++m) {
      const u16* ap = SWt + ((size_t)(bh * NGRP + m * 16 + lrow)) * NVOX + n0 + (hi ? 8 : 0);
      af[m].q[0] = ld16(ap);
      af[m].q[1] = ld16(ap + 16);
    }
    FragBF bf[4];
    #pragma unroll
    for (int c = 0; c < 4; ++c) {
      const u16* bp = XPt + ((size_t)(bh * DHEAD + c * 16 + lrow)) * NVOX + n0 + (hi ? 8 : 0);
      bf[c].q[0] = ld16(bp);
      bf[c].q[1] = ld16(bp + 16);
    }
    #pragma unroll
    for (int m = 0; m < 2; ++m)
      #pragma unroll
      for (int c = 0; c < 4; ++c)
        acc[m][c] = wmma_bf16(af[m], bf[c], acc[m][c]);
  }
  #pragma unroll
  for (int m = 0; m < 2; ++m)
    #pragma unroll
    for (int c = 0; c < 4; ++c)
      #pragma unroll
      for (int e = 0; e < 8; ++e) {
        int g  = m * 16 + hi * 8 + e;
        int cc = c * 16 + lrow;
        atomicAdd(&EID[((size_t)bh * NGRP + g) * DHEAD + cc], acc[m][c][e]);
      }
}

// --------------------------- K10: eid finalize -----------------------------
// EIDb[bh][c][g] = bf16(EID[bh][g][c] / (1 + 1e-5))
__global__ void k_eid_fin(const float* __restrict__ EID, u16* __restrict__ EIDb) {
  int i = blockIdx.x * 256 + threadIdx.x;   // < 32*32*64
  int g  = i & 31;
  int c  = (i >> 5) & 63;
  int bh = i >> 11;
  EIDb[((size_t)bh * DHEAD + c) * NGRP + g] =
      f2bf(EID[((size_t)bh * NGRP + g) * DHEAD + c] * (1.0f / 1.00001f));
}

// ------------- K11: fused out_x = sw@eid, out = out_x@out_w^T + b ----------
// grid (512 n-tiles, 4 batch), 256 threads (8 waves). LDS holds 64x512 bf16.
__global__ void k_final(const u16* __restrict__ SW, const u16* __restrict__ EIDb,
                        const u16* __restrict__ OWb, const float* __restrict__ out_b,
                        float* __restrict__ out) {
  __shared__ __align__(16) u16 ox[64 * INNER];
  const int nt = blockIdx.x, bb = blockIdx.y;
  const int n0 = nt * 64;
  const int t = threadIdx.x, lane = t & 31, wv = t >> 5;
  const int lrow = lane & 15, hi = lane >> 4;

  // ---- stage 1: wave wv computes head wv: [64 x 32] @ [32 x 64] ----
  {
    const int hg = wv;
    const size_t bh = (size_t)bb * NHEAD + hg;
    FragBF bf[4];
    #pragma unroll
    for (int c = 0; c < 4; ++c) {
      const u16* bp = EIDb + (bh * DHEAD + c * 16 + lrow) * NGRP + (hi ? 8 : 0);
      bf[c].q[0] = ld16(bp);
      bf[c].q[1] = ld16(bp + 16);
    }
    for (int m = 0; m < 4; ++m) {
      FragBF af;
      const u16* ap = SW + (bh * NVOX + n0 + m * 16 + lrow) * NGRP + (hi ? 8 : 0);
      af.q[0] = ld16(ap);
      af.q[1] = ld16(ap + 16);
      #pragma unroll
      for (int c = 0; c < 4; ++c) {
        v8f acc = {};
        acc = wmma_bf16(af, bf[c], acc);
        #pragma unroll
        for (int e = 0; e < 8; ++e) {
          int row = m * 16 + hi * 8 + e;
          int col = hg * DHEAD + c * 16 + lrow;
          ox[row * INNER + col] = f2bf(acc[e]);
        }
      }
    }
  }
  __syncthreads();

  // ---- stage 2: [64 x 512] @ [512 x 256]; waves 4(M) x 2(N of 128) ----
  {
    const int wm = wv & 3, wn = wv >> 2;
    v8f acc[8] = {};
    for (int cib = 0; cib < 16; ++cib) {
      FragBF af;
      const u16* ap = ox + (wm * 16 + lrow) * INNER + cib * 32 + (hi ? 8 : 0);
      af.q[0] = ld16(ap);
      af.q[1] = ld16(ap + 16);
      #pragma unroll
      for (int nn = 0; nn < 8; ++nn) {
        FragBF bf;
        const u16* bp = OWb + ((size_t)cib * CDIM + wn * 128 + nn * 16 + lrow) * 32 + (hi ? 8 : 0);
        bf.q[0] = ld16(bp);
        bf.q[1] = ld16(bp + 16);
        acc[nn] = wmma_bf16(af, bf, acc[nn]);
      }
    }
    #pragma unroll
    for (int nn = 0; nn < 8; ++nn) {
      int col = wn * 128 + nn * 16 + lrow;
      float bias = out_b[col];
      #pragma unroll
      for (int e = 0; e < 8; ++e) {
        int row = wm * 16 + hi * 8 + e;
        out[((size_t)bb * NVOX + n0 + row) * CDIM + col] = acc[nn][e] + bias;
      }
    }
  }
}

// ---------------------------------------------------------------------------
extern "C" void kernel_launch(void* const* d_in, const int* in_sizes, int n_in,
                              void* d_out, int out_size, void* d_ws, size_t ws_size,
                              hipStream_t stream) {
  (void)in_sizes; (void)n_in; (void)out_size; (void)ws_size;
  const float* x       = (const float*)d_in[0];
  const float* conv_w  = (const float*)d_in[1];
  const float* conv_b  = (const float*)d_in[2];
  const float* slice_w = (const float*)d_in[3];
  const float* slice_b = (const float*)d_in[4];
  const float* ada_w   = (const float*)d_in[5];
  const float* ada_b   = (const float*)d_in[6];
  const float* out_w   = (const float*)d_in[7];
  const float* out_b   = (const float*)d_in[8];
  float* out = (float*)d_out;

  char* ws = (char*)d_ws;
  size_t off = 0;
  auto alloc = [&](size_t bytes) -> void* {
    void* p = ws + off;
    off = (off + bytes + 255) & ~(size_t)255;
    return p;
  };
  u16*      Xb   = (u16*)alloc((size_t)NB * NVOX * CDIM * 2);        // 64 MB
  u16*      Wb   = (u16*)alloc((size_t)27 * CDIM * INNER * 2);       // 6.75 MB
  u16*      OWb  = (u16*)alloc((size_t)INNER * CDIM * 2);            // 256 KB
  u16*      XPb  = (u16*)alloc((size_t)NB * NVOX * INNER * 2);       // 128 MB
  u16*      XPt  = (u16*)alloc((size_t)NB * NVOX * INNER * 2);       // 128 MB
  float*    Z    = (float*)alloc((size_t)NB * NHEAD * NVOX * NGRP * 4); // 128 MB
  unsigned* GMAX = (unsigned*)alloc((size_t)NB * NHEAD * NGRP * 4);
  float*    GSUM = (float*)alloc((size_t)NB * NHEAD * NGRP * 4);
  u16*      SW   = (u16*)alloc((size_t)NB * NHEAD * NVOX * NGRP * 2);   // 64 MB
  u16*      SWt  = (u16*)alloc((size_t)NB * NHEAD * NVOX * NGRP * 2);   // 64 MB
  float*    EID  = (float*)alloc((size_t)NB * NHEAD * NGRP * DHEAD * 4);
  u16*      EIDb = (u16*)alloc((size_t)NB * NHEAD * NGRP * DHEAD * 2);

  k_cast_x  <<<32768, 256, 0, stream>>>(x, Xb);
  k_pack_w  <<<13824, 256, 0, stream>>>(conv_w, Wb);
  k_pack_outw<<<512,  256, 0, stream>>>(out_w, OWb);
  k_init    <<<256,   256, 0, stream>>>(GMAX, GSUM, EID);
  k_conv    <<<dim3(1024, 8), 256, 0, stream>>>(Xb, Wb, conv_b, XPb, XPt);
  k_logits  <<<dim3(128, NHEAD, NB), 256, 0, stream>>>(XPb, slice_w, slice_b,
                                                       ada_w, ada_b, Z, GMAX);
  k_sum     <<<dim3(128, NHEAD, NB), 256, 0, stream>>>(Z, GMAX, GSUM);
  k_weights <<<dim3(128, NHEAD, NB), 256, 0, stream>>>(Z, GMAX, GSUM, SW, SWt);
  k_eid     <<<dim3(32, 32), 32, 0, stream>>>(SWt, XPt, EID);
  k_eid_fin <<<256, 256, 0, stream>>>(EID, EIDb);
  k_final   <<<dim3(512, NB), 256, 0, stream>>>(SW, EIDb, OWb, out_b, out);
}
